// MultiScaleDeformableAttention_74663711474023
// MI455X (gfx1250) — matrix operand order repaired and checked by reading, not compiled, not run on hardware
//
#include <hip/hip_runtime.h>
#include <hip/hip_bf16.h>

// ---------------------------------------------------------------------------
// Multi-Scale Deformable Attention for MI455X (gfx1250, wave32, WMMA).
// GEMMs use v_wmma_f32_16x16x32_f16 with an error-compensated f16 hi/lo split
// (3 WMMAs per k-step, ~2^-21 rel. error vs fp32). Operands are pre-packed
// once into the WMMA per-lane fragment layout so the GEMM inner loop is pure
// b128 loads + WMMA (no per-tile conversion VALU).
// ---------------------------------------------------------------------------

typedef __attribute__((ext_vector_type(16))) _Float16 v16h;
typedef __attribute__((ext_vector_type(8)))  float    v8f;

#define BS      2
#define NQ      7681
#define NV      7681
#define CDIM    256
#define MHEADS  8
#define DHEAD   32
#define MROWS   (BS * NQ)          // 15362
#define KDIM    256
#define KBLK    (KDIM / 32)        // 8 k-steps of 32
#define TILES_M ((MROWS + 15) / 16) // 961

// Packed fragment layout: [frag = tile*KBLK + kb][lane 0..31][16 halfs]
// Each lane's 16 halfs are contiguous (32 B) -> two global_load_b128.

// -------------------- Pack A (M-side) into f16 hi/lo fragments -------------
// A fragment (16x32 f16): lane row = lane&15, half = lane>>4;
//   element e<8 : K = kb*32 + 8*half + e
//   element e>=8: K = kb*32 + 16 + 8*half + (e-8)
__global__ void pack_a_kernel(const float* __restrict__ A,
                              _Float16* __restrict__ Ah,
                              _Float16* __restrict__ Al,
                              int Mrows, int nfrags) {
  const int t = (int)(blockIdx.x * blockDim.x + threadIdx.x);
  const int frag = t >> 5;
  const int lane = t & 31;
  if (frag >= nfrags) return;
  const int tm   = frag / KBLK;
  const int kb   = frag % KBLK;
  const int half = lane >> 4;
  const int lrow = tm * 16 + (lane & 15);
  const int arow = (lrow < Mrows) ? lrow : (Mrows - 1);

  const float* ap = A + (size_t)arow * KDIM + kb * 32 + 8 * half;
  _Float16* dh = Ah + ((size_t)frag * 32 + lane) * 16;
  _Float16* dl = Al + ((size_t)frag * 32 + lane) * 16;
#pragma unroll
  for (int i = 0; i < 8; ++i) {
    const float v0 = ap[i];
    const float v1 = ap[16 + i];
    const _Float16 h0 = (_Float16)v0;
    const _Float16 h1 = (_Float16)v1;
    dh[i]     = h0;  dl[i]     = (_Float16)(v0 - (float)h0);
    dh[8 + i] = h1;  dl[8 + i] = (_Float16)(v1 - (float)h1);
  }
}

// -------------------- Pack B (N-side) into f16 hi/lo fragments -------------
// B fragment (32x16 f16): lane col = lane&15, half = lane>>4;
//   element e<8 : K = kb*32 + 8*half + e
//   element e>=8: K = kb*32 + 16 + 8*half + (e-8)
__global__ void pack_b_kernel(const float* __restrict__ B,
                              _Float16* __restrict__ Bh,
                              _Float16* __restrict__ Bl,
                              int N, int nfrags) {
  const int t = (int)(blockIdx.x * blockDim.x + threadIdx.x);
  const int frag = t >> 5;
  const int lane = t & 31;
  if (frag >= nfrags) return;
  const int tn   = frag / KBLK;
  const int kb   = frag % KBLK;
  const int half = lane >> 4;
  const int col  = tn * 16 + (lane & 15);
  const int k0   = kb * 32 + 8 * half;

  const float* bp = B + (size_t)k0 * N + col;
  _Float16* dh = Bh + ((size_t)frag * 32 + lane) * 16;
  _Float16* dl = Bl + ((size_t)frag * 32 + lane) * 16;
#pragma unroll
  for (int j = 0; j < 16; ++j) {
    const float v = bp[(size_t)(j + (j >= 8 ? 8 : 0)) * N];
    const _Float16 h = (_Float16)v;
    dh[j] = h;
    dl[j] = (_Float16)(v - (float)h);
  }
}

// -------------------- WMMA GEMM on pre-packed fragments --------------------
// One wave per 16x16 tile; inner loop: 8x b128 loads + 3 WMMAs per k-step.
__global__ void wmma_gemm_packed_kernel(const v16h* __restrict__ Ah,
                                        const v16h* __restrict__ Al,
                                        const v16h* __restrict__ Bh,
                                        const v16h* __restrict__ Bl,
                                        const float* __restrict__ bias,
                                        float* __restrict__ C,
                                        int Mrows, int N,
                                        int tiles_m, int tiles_n) {
  const int wave  = (int)((blockIdx.x * blockDim.x + threadIdx.x) >> 5);
  const int lane  = (int)(threadIdx.x & 31);
  const int total = tiles_m * tiles_n;
  if (wave >= total) return;                 // wave-uniform: EXEC stays all-1s

  const int tm = wave / tiles_n;
  const int tn = wave % tiles_n;

  const size_t abase = ((size_t)tm * KBLK) * 32 + lane;  // v16h units
  const size_t bbase = ((size_t)tn * KBLK) * 32 + lane;

  v8f acc = {};
#pragma unroll
  for (int kb = 0; kb < KBLK; ++kb) {
    const v16h ah = Ah[abase + (size_t)kb * 32];
    const v16h al = Al[abase + (size_t)kb * 32];
    const v16h bh = Bh[bbase + (size_t)kb * 32];
    const v16h bl = Bl[bbase + (size_t)kb * 32];
    // C += ah*bh + al*bh + ah*bl
    acc = __builtin_amdgcn_wmma_f32_16x16x32_f16(false, ah, false, bh,
                                                 (short)0, acc, false, false);
    acc = __builtin_amdgcn_wmma_f32_16x16x32_f16(false, al, false, bh,
                                                 (short)0, acc, false, false);
    acc = __builtin_amdgcn_wmma_f32_16x16x32_f16(false, ah, false, bl,
                                                 (short)0, acc, false, false);
  }

  const int half = lane >> 4;
  const int col  = tn * 16 + (lane & 15);
  const float bb = bias ? bias[col] : 0.0f;
  // C/D layout: VGPR i -> row tm*16 + 8*half + i, column `col`.
  float* cp = C + (size_t)(tm * 16 + 8 * half) * N + col;
  if ((tm * 16 + 16) <= Mrows) {
    // Fast path (all but the single tail tile): unguarded strided stores.
#pragma unroll
    for (int i = 0; i < 8; ++i) cp[(size_t)i * N] = acc[i] + bb;
  } else {
#pragma unroll
    for (int i = 0; i < 8; ++i) {
      const int r = tm * 16 + 8 * half + i;
      if (r < Mrows) cp[(size_t)i * N] = acc[i] + bb;
    }
  }
}

// -------------------- Sampling: softmax + bilinear gather ------------------
// One wave per (b, q, head); lane = channel d (DHEAD == 32 == wave32).
__global__ void msda_sample_kernel(const float* __restrict__ refp,
                                   const float* __restrict__ vproj,
                                   const float* __restrict__ offb,
                                   const float* __restrict__ attnb,
                                   float* __restrict__ tmp) {
  const int wave = (int)((blockIdx.x * blockDim.x + threadIdx.x) >> 5);
  const int lane = (int)(threadIdx.x & 31);              // channel d
  const int total = BS * NQ * MHEADS;
  if (wave >= total) return;

  const int m  = wave & (MHEADS - 1);
  const int bq = wave >> 3;                               // b*NQ + q
  const int b  = bq / NQ;

  const float rx = refp[bq * 2 + 0];
  const float ry = refp[bq * 2 + 1];
  const float* off = offb  + (size_t)bq * 256 + m * 32;   // [L][K][2]
  const float* lg  = attnb + (size_t)bq * 128 + m * 16;   // 16 logits

  // Softmax over the 16 logits (broadcast loads; redundant across lanes).
  float p[16];
  float mx = -1e30f;
#pragma unroll
  for (int i = 0; i < 16; ++i) { p[i] = lg[i]; mx = fmaxf(mx, p[i]); }
  float s = 0.0f;
#pragma unroll
  for (int i = 0; i < 16; ++i) { p[i] = __expf(p[i] - mx); s += p[i]; }
  const float inv = 1.0f / s;

  const int lvlH[4] = {76, 38, 19, 10};
  const int lvlW[4] = {76, 38, 19, 10};
  const int lvlS[4] = {0, 5776, 7220, 7581};

  // vproj layout: [b][nv][m][d] == [b][nv][C] with c = m*32 + d
  const float* vbase = vproj + ((size_t)b * NV) * CDIM + m * DHEAD + lane;

  float acc = 0.0f;
#pragma unroll
  for (int l = 0; l < 4; ++l) {
    const int H = lvlH[l], W = lvlW[l], S = lvlS[l];
#pragma unroll
    for (int k = 0; k < 4; ++k) {
      const float ox = off[(l * 4 + k) * 2 + 0];
      const float oy = off[(l * 4 + k) * 2 + 1];
      const float lx = fminf(fmaxf(rx + ox, 0.0f), 1.0f);
      const float ly = fminf(fmaxf(ry + oy, 0.0f), 1.0f);
      // align_corners=False unnormalize: x = loc*W - 0.5
      const float x = lx * (float)W - 0.5f;
      const float y = ly * (float)H - 0.5f;
      const float xf = floorf(x), yf = floorf(y);
      const int   x0 = (int)xf,  y0 = (int)yf;
      const float wx1 = x - xf,  wy1 = y - yf;
      const float wx0 = 1.0f - wx1, wy0 = 1.0f - wy1;

      float sample = 0.0f;
      // Branchless corners: clamp index, zero the weight when out-of-range
      // (identical semantics to the reference's padding_mode='zeros').
      auto corner = [&](int xx, int yy, float wc) {
        const bool valid = (xx >= 0) & (xx < W) & (yy >= 0) & (yy < H);
        const int cx = xx < 0 ? 0 : (xx >= W ? W - 1 : xx);
        const int cy = yy < 0 ? 0 : (yy >= H ? H - 1 : yy);
        const float wv = valid ? wc : 0.0f;
        sample += wv * vbase[(size_t)(S + cy * W + cx) * CDIM];
      };
      corner(x0,     y0,     wx0 * wy0);
      corner(x0 + 1, y0,     wx1 * wy0);
      corner(x0,     y0 + 1, wx0 * wy1);
      corner(x0 + 1, y0 + 1, wx1 * wy1);

      // Faithful source quirk: sample (level l, point k) pairs with
      // softmax weight at flat index k*4 + l.
      acc += (p[k * 4 + l] * inv) * sample;
    }
  }

  tmp[(size_t)bq * CDIM + m * DHEAD + lane] = acc;
}

// ---------------------------------------------------------------------------
extern "C" void kernel_launch(void* const* d_in, const int* in_sizes, int n_in,
                              void* d_out, int out_size, void* d_ws, size_t ws_size,
                              hipStream_t stream) {
  const float* query  = (const float*)d_in[0];
  const float* refp   = (const float*)d_in[1];
  const float* value  = (const float*)d_in[2];
  // d_in[3]: value_level_start_index (int32) — level geometry is compile-time.
  const float* W_off  = (const float*)d_in[4];
  const float* b_off  = (const float*)d_in[5];
  const float* W_attn = (const float*)d_in[6];
  const float* b_attn = (const float*)d_in[7];
  const float* W_val  = (const float*)d_in[8];
  const float* b_val  = (const float*)d_in[9];
  const float* W_out  = (const float*)d_in[10];
  const float* b_out  = (const float*)d_in[11];
  float* out = (float*)d_out;

  // -------- Workspace carve-out --------
  const size_t afrags = (size_t)TILES_M * KBLK;        // 7688 fragments
  const size_t bfrags = (size_t)16 * KBLK;             // up to N=256 -> 128
  float* ws    = (float*)d_ws;
  float* vproj = ws;                                   // [15362, 256]
  float* offb  = vproj + (size_t)MROWS * 256;          // [15362, 256]
  float* attnb = offb  + (size_t)MROWS * 256;          // [15362, 128]
  float* tmp   = attnb + (size_t)MROWS * 128;          // [15362, 256]
  _Float16* Ah = (_Float16*)(tmp + (size_t)MROWS * 256);
  _Float16* Al = Ah + afrags * 32 * 16;                // ~3.9 MF16 each
  _Float16* Bh = Al + afrags * 32 * 16;
  _Float16* Bl = Bh + bfrags * 32 * 16;

  auto pack_a = [&](const float* A) {
    const int nf = (int)afrags;
    const int threads = nf * 32;
    pack_a_kernel<<<(threads + 255) / 256, 256, 0, stream>>>(A, Ah, Al,
                                                             MROWS, nf);
  };
  auto pack_b = [&](const float* B, int N) {
    const int nf = (N / 16) * KBLK;
    const int threads = nf * 32;
    pack_b_kernel<<<(threads + 255) / 256, 256, 0, stream>>>(B, Bh, Bl, N, nf);
  };
  auto gemm = [&](const float* bias, float* C, int N) {
    const int tn = N / 16;
    const int tiles = TILES_M * tn;
    const int blocks = (tiles + 7) / 8;                // 8 waves / 256-thread block
    wmma_gemm_packed_kernel<<<blocks, 256, 0, stream>>>(
        (const v16h*)Ah, (const v16h*)Al, (const v16h*)Bh, (const v16h*)Bl,
        bias, C, MROWS, N, TILES_M, tn);
  };

  // 1) value projection: vproj = value @ W_val + b_val
  pack_a(value);
  pack_b(W_val, 256);
  gemm(b_val, vproj, 256);

  // 2) offsets + attention logits from query (A-pack reused for both)
  pack_a(query);
  pack_b(W_off, 256);
  gemm(b_off, offb, 256);
  pack_b(W_attn, 128);
  gemm(b_attn, attnb, 128);

  // 3) deformable sampling -> tmp
  {
    const int waves  = BS * NQ * MHEADS;               // 122896
    const int blocks = (waves + 7) / 8;
    msda_sample_kernel<<<blocks, 256, 0, stream>>>(refp, vproj, offb, attnb, tmp);
  }

  // 4) output projection: out = tmp @ W_out + b_out
  pack_a(tmp);
  pack_b(W_out, 256);
  gemm(b_out, out, 256);
}
